// MultiHeadedAttention_15625091023209
// MI455X (gfx1250) — compile-verified, matrix operand order checked by reading, not run on hardware
//
#include <hip/hip_runtime.h>
#include <hip/hip_bf16.h>

// ---------- types ----------
typedef unsigned short u16;
typedef __attribute__((ext_vector_type(16))) __bf16 v16bf;
typedef __attribute__((ext_vector_type(8)))  float  v8f;
typedef __attribute__((ext_vector_type(4)))  unsigned int v4u;
typedef __attribute__((ext_vector_type(8)))  int v8i;
typedef __attribute__((ext_vector_type(4)))  int v4i;

#define HAVE_TDM __has_builtin(__builtin_amdgcn_tensor_load_to_lds)

// fp32 -> bf16 (round-to-nearest-even)
__device__ __forceinline__ u16 f2bf(float x) {
  union { float f; unsigned int u; } c; c.f = x;
  unsigned int r = c.u + 0x7FFFu + ((c.u >> 16) & 1u);
  return (u16)(r >> 16);
}

// Load a 16-element bf16 fragment as two 16-byte chunks.
// For both A (16x32, row per lane) and B (32x16, column per lane) layouts,
// lane l<16 holds K {0..7,16..23}, lane l>=16 holds K {8..15,24..31}:
// caller passes p0 = base + half*8, p1 = base + 16 + half*8 (half = lane>>4).
__device__ __forceinline__ v16bf load_frag(const u16* p0, const u16* p1) {
  union { v16bf v; v4u q[2]; } f;
  f.q[0] = *reinterpret_cast<const v4u*>(p0);
  f.q[1] = *reinterpret_cast<const v4u*>(p1);
  return f.v;
}

#if HAVE_TDM
// Tensor Data Mover: 2D tile (bf16) global -> LDS.
// D# bitfields per CDNA5 ISA.
// 6-arg builtin form: (g0, g1, g2, g3, g_extra, cpol).
__device__ __forceinline__ void tdm_load_2d_bf16(
    unsigned lds_off, const u16* gptr,
    unsigned tile_d0, unsigned tile_d1,       // tile extent (elements)
    unsigned tensor_d0, unsigned tensor_d1,   // tensor extent (elements)
    unsigned long long stride0)               // dim0 stride (elements)
{
  unsigned long long ga = (unsigned long long)(size_t)gptr;
  v4u g0;
  g0[0] = 1u;                                        // count=1, user mode, no gather
  g0[1] = lds_off;                                   // lds_addr (bytes)
  g0[2] = (unsigned)(ga & 0xFFFFFFFFu);              // global_addr[31:0]
  g0[3] = (unsigned)((ga >> 32) & 0x01FFFFFFu)       // global_addr[56:32]
          | (2u << 30);                              // type = 2 ("image")
  v8i g1;
  g1[0] = (int)(1u << 16);                           // data_size=1 (2B), wg_mask=0
  g1[1] = (int)((tensor_d0 & 0xFFFFu) << 16);        // tensor_dim0[15:0]
  g1[2] = (int)(((tensor_d0 >> 16) & 0xFFFFu) |      // tensor_dim0[31:16]
                ((tensor_d1 & 0xFFFFu) << 16));      // tensor_dim1[15:0]
  g1[3] = (int)(((tensor_d1 >> 16) & 0xFFFFu) |      // tensor_dim1[31:16]
                ((tile_d0 & 0xFFFFu) << 16));        // tile_dim0
  g1[4] = (int)(tile_d1 & 0xFFFFu);                  // tile_dim1 (tile_dim2=0)
  g1[5] = (int)(unsigned)(stride0 & 0xFFFFFFFFu);    // tensor_dim0_stride[31:0]
  g1[6] = (int)(unsigned)((stride0 >> 32) & 0xFFFFu);// stride[47:32], dim1_stride lo=0
  g1[7] = 0;
  v4i z4 = {0, 0, 0, 0};
  v8i z8 = {0, 0, 0, 0, 0, 0, 0, 0};
  __builtin_amdgcn_tensor_load_to_lds(g0, g1, z4, z4, z8, 0);
}
#endif

__device__ __forceinline__ void wait_tensorcnt0() {
#if __has_builtin(__builtin_amdgcn_s_wait_tensorcnt)
  __builtin_amdgcn_s_wait_tensorcnt(0);
#else
  asm volatile("s_wait_tensorcnt 0x0" ::: "memory");
#endif
}

// ---------- conversion kernels ----------
__global__ void cvt_bf16_kernel(const float* __restrict__ src, u16* __restrict__ dst, int n) {
  for (int i = blockIdx.x * blockDim.x + threadIdx.x; i < n; i += gridDim.x * blockDim.x)
    dst[i] = f2bf(src[i]);
}

// src[K][N] (row-major f32) -> dst[N][K] (row-major bf16)
__global__ void transp_bf16_kernel(const float* __restrict__ src, u16* __restrict__ dst,
                                   int K, int N) {
  int total = K * N;
  for (int i = blockIdx.x * blockDim.x + threadIdx.x; i < total; i += gridDim.x * blockDim.x) {
    int k = i / N, n = i - k * N;
    dst[n * K + k] = f2bf(src[i]);
  }
}

// ---------- WMMA GEMM:  C[M][Nout] = A[M][K] * Bt[Nout][K]^T + bias ----------
// MODE 0: QKV epilogue -> scatter Q[BH][N][64], K[BH][N][64], Vt[BH][64][N] (bf16)
// MODE 1: plain f32 output [M][Nout]
template <int MODE>
__global__ __launch_bounds__(256) void gemm_bf16_kernel(
    const u16* __restrict__ A, const u16* __restrict__ Bt,
    const float* __restrict__ bias,
    float* __restrict__ outF,
    u16* __restrict__ qOut, u16* __restrict__ kOut, u16* __restrict__ vtOut,
    int M, int K, int Nout)
{
  __shared__ __attribute__((aligned(16))) u16 As[64 * 32];
  __shared__ __attribute__((aligned(16))) u16 Bs[128 * 32];

  const int tid  = threadIdx.x;
  const int wave = tid >> 5, lane = tid & 31;
  const int half = lane >> 4, l16 = lane & 15;

  const int nTiles = Nout >> 7;                 // Nout / 128
  const int mBase = (blockIdx.x / nTiles) * 64;
  const int nBase = (blockIdx.x % nTiles) * 128;
  const int mSub = (wave & 1) * 32;
  const int nSub = (wave >> 1) * 32;

  v8f acc[2][2] = {};

  const int arow = tid >> 2, akc = (tid & 3) << 3;   // A tile: 64 rows x 32 k, 8 bf16/thread
  const int brow = tid >> 1, bkc = (tid & 1) << 4;   // B tile: 128 rows x 32 k, 16 bf16/thread

  for (int k0 = 0; k0 < K; k0 += 32) {
    *(v4u*)&As[arow * 32 + akc] = *(const v4u*)&A[(mBase + arow) * K + k0 + akc];
    *(v4u*)&Bs[brow * 32 + bkc]     = *(const v4u*)&Bt[(nBase + brow) * K + k0 + bkc];
    *(v4u*)&Bs[brow * 32 + bkc + 8] = *(const v4u*)&Bt[(nBase + brow) * K + k0 + bkc + 8];
    // speculative prefetch of next K tile (global_prefetch_b8)
    __builtin_prefetch(&A[(mBase + arow) * K + k0 + 32 + akc], 0, 0);
    __builtin_prefetch(&Bt[(nBase + brow) * K + k0 + 32 + bkc], 0, 0);
    __syncthreads();

    v16bf af[2], bfr[2];
#pragma unroll
    for (int mt = 0; mt < 2; ++mt) {
      const u16* b = &As[(mSub + mt * 16 + l16) * 32];
      af[mt] = load_frag(b + half * 8, b + 16 + half * 8);
    }
#pragma unroll
    for (int nt = 0; nt < 2; ++nt) {
      const u16* b = &Bs[(nSub + nt * 16 + l16) * 32];
      bfr[nt] = load_frag(b + half * 8, b + 16 + half * 8);
    }
#pragma unroll
    for (int mt = 0; mt < 2; ++mt)
#pragma unroll
      for (int nt = 0; nt < 2; ++nt)
        acc[mt][nt] = __builtin_amdgcn_wmma_f32_16x16x32_bf16(
            false, af[mt], false, bfr[nt], (short)0, acc[mt][nt], false, false);
    __syncthreads();
  }

  // epilogue (C layout: VGPR r -> row r + 8*half, col = l16)
#pragma unroll
  for (int mt = 0; mt < 2; ++mt)
#pragma unroll
    for (int nt = 0; nt < 2; ++nt) {
      const int gcol = nBase + nSub + nt * 16 + l16;
      const float bv = bias[gcol];
#pragma unroll
      for (int r = 0; r < 8; ++r) {
        const int gm = mBase + mSub + mt * 16 + r + half * 8;
        const float val = acc[mt][nt][r] + bv;
        if (MODE == 1) {
          outF[gm * Nout + gcol] = val;
        } else {
          const int sect = gcol >> 10, c = gcol & 1023;
          const int h = c >> 6, hd = c & 63;
          const int b = gm >> 11, n = gm & 2047;
          const int bh = b * 16 + h;
          const u16 bfv = f2bf(val);
          if (sect == 0)      qOut[(bh * 2048 + n) * 64 + hd] = bfv;
          else if (sect == 1) kOut[(bh * 2048 + n) * 64 + hd] = bfv;
          else                vtOut[(bh * 64 + hd) * 2048 + n] = bfv;
        }
      }
    }
}

// ---------- flash attention ----------
// 1 workgroup = 128 query rows of one head; 1 wave = 16 rows.
// Per 64-key chunk: K tile (64x64) and V tile (64x64) staged into LDS once per
// workgroup via the Tensor Data Mover, shared by all 8 waves.
__global__ __launch_bounds__(256) void flash_attn_kernel(
    const u16* __restrict__ Q, const u16* __restrict__ Km,
    const u16* __restrict__ Vt, u16* __restrict__ ctx)
{
  __shared__ __attribute__((aligned(16))) u16 Ks[64 * 64];      // [kv_local][d]
  __shared__ __attribute__((aligned(16))) u16 Vs[64 * 64];      // [d][kv_local]
  __shared__ __attribute__((aligned(16))) u16 Pl[8][16 * 64];   // per-wave P tile

  const int tid = threadIdx.x, wave = tid >> 5, lane = tid & 31;
  const int half = lane >> 4, l16 = lane & 15;

  const int nQblk = 2048 / 128;                  // 16 query blocks per head
  const int bh   = blockIdx.x / nQblk;
  const int qblk = blockIdx.x % nQblk;
  const int qbase = qblk * 128 + wave * 16;

  const u16* Qp = Q  + bh * 2048 * 64;
  const u16* Kp = Km + bh * 2048 * 64;
  const u16* Vp = Vt + bh * 64 * 2048;

  // Q fragments (A operand, registers for the whole kv loop)
  v16bf aq[2];
  {
    const u16* b0 = Qp + (qbase + l16) * 64;
    aq[0] = load_frag(b0 + half * 8,      b0 + 16 + half * 8);
    aq[1] = load_frag(b0 + 32 + half * 8, b0 + 48 + half * 8);
  }

  float mrow[8], lrow[8];
  v8f oacc[4] = {};
#pragma unroll
  for (int r = 0; r < 8; ++r) { mrow[r] = -1e30f; lrow[r] = 0.f; }

  const float SCL = 0.125f;          // 1/sqrt(64)
  const float L2E = 1.44269504f;     // log2(e)

  for (int kv = 0; kv < 2048; kv += 64) {
    // ---- stage K/V tiles into LDS ----
#if HAVE_TDM
    if (wave == 0) {
      tdm_load_2d_bf16((unsigned)(size_t)(void*)Ks, Kp + kv * 64,
                       /*tile*/ 64, 64, /*tensor*/ 64, 2048, /*stride0*/ 64);
      tdm_load_2d_bf16((unsigned)(size_t)(void*)Vs, Vp + kv,
                       /*tile*/ 64, 64, /*tensor*/ 2048, 64, /*stride0*/ 2048);
      wait_tensorcnt0();
    }
#else
    {
      const int r = tid >> 2, c = (tid & 3) << 4;  // 64 rows x 64 elems, 16/thread
      *(v4u*)&Ks[r * 64 + c]     = *(const v4u*)&Kp[(kv + r) * 64 + c];
      *(v4u*)&Ks[r * 64 + c + 8] = *(const v4u*)&Kp[(kv + r) * 64 + c + 8];
      *(v4u*)&Vs[r * 64 + c]     = *(const v4u*)&Vp[r * 2048 + kv + c];
      *(v4u*)&Vs[r * 64 + c + 8] = *(const v4u*)&Vp[r * 2048 + kv + c + 8];
    }
#endif
    __syncthreads();

    // ---- scores: four 16x16 tiles over this 64-key chunk ----
    v8f s[4] = {};
#pragma unroll
    for (int t = 0; t < 4; ++t) {
      const u16* kb = &Ks[(t * 16 + l16) * 64];
      v16bf bk0 = load_frag(kb + half * 8,      kb + 16 + half * 8);
      v16bf bk1 = load_frag(kb + 32 + half * 8, kb + 48 + half * 8);
      s[t] = __builtin_amdgcn_wmma_f32_16x16x32_bf16(false, aq[0], false, bk0, (short)0, s[t], false, false);
      s[t] = __builtin_amdgcn_wmma_f32_16x16x32_bf16(false, aq[1], false, bk1, (short)0, s[t], false, false);
    }

    // ---- online softmax (rows split across 16-lane halves) ----
    float alpha[8];
#pragma unroll
    for (int r = 0; r < 8; ++r) {
      float v = fmaxf(fmaxf(s[0][r], s[1][r]), fmaxf(s[2][r], s[3][r])) * SCL;
#pragma unroll
      for (int m = 8; m >= 1; m >>= 1) v = fmaxf(v, __shfl_xor(v, m, 16));
      const float mnew = fmaxf(mrow[r], v);
      alpha[r] = exp2f((mrow[r] - mnew) * L2E);
      float p[4], ps = 0.f;
#pragma unroll
      for (int t = 0; t < 4; ++t) {
        p[t] = exp2f((s[t][r] * SCL - mnew) * L2E);
        ps += p[t];
      }
#pragma unroll
      for (int m = 8; m >= 1; m >>= 1) ps += __shfl_xor(ps, m, 16);
      lrow[r] = lrow[r] * alpha[r] + ps;
      mrow[r] = mnew;
      const int prow = r + half * 8;
#pragma unroll
      for (int t = 0; t < 4; ++t)
        Pl[wave][prow * 64 + t * 16 + l16] = f2bf(p[t]);
    }
    __builtin_amdgcn_wave_barrier();   // DS stores before DS loads (same wave)

    // re-load P as two A fragments (kv 0..31, 32..63)
    v16bf ap0, ap1;
    {
      const u16* pb = &Pl[wave][l16 * 64];
      ap0 = load_frag(pb + half * 8,      pb + 16 + half * 8);
      ap1 = load_frag(pb + 32 + half * 8, pb + 48 + half * 8);
    }

    // ---- O = O*alpha + P @ V ----
#pragma unroll
    for (int dt = 0; dt < 4; ++dt) {
#pragma unroll
      for (int r = 0; r < 8; ++r) oacc[dt][r] *= alpha[r];
      const u16* vb = &Vs[(dt * 16 + l16) * 64];
      v16bf bv0 = load_frag(vb + half * 8,      vb + 16 + half * 8);
      v16bf bv1 = load_frag(vb + 32 + half * 8, vb + 48 + half * 8);
      oacc[dt] = __builtin_amdgcn_wmma_f32_16x16x32_bf16(false, ap0, false, bv0, (short)0, oacc[dt], false, false);
      oacc[dt] = __builtin_amdgcn_wmma_f32_16x16x32_bf16(false, ap1, false, bv1, (short)0, oacc[dt], false, false);
    }
    __syncthreads();   // protect Ks/Vs before next chunk's staging
  }

  // normalize and store context [B][N][H][Hd] as bf16
  const int b = bh >> 4, h = bh & 15;
#pragma unroll
  for (int dt = 0; dt < 4; ++dt) {
    const int dcol = dt * 16 + l16;
#pragma unroll
    for (int r = 0; r < 8; ++r) {
      const int n = qbase + r + half * 8;
      ctx[((b * 2048 + n) * 16 + h) * 64 + dcol] = f2bf(oacc[dt][r] / lrow[r]);
    }
  }
}

// ---------- launcher ----------
extern "C" void kernel_launch(void* const* d_in, const int* in_sizes, int n_in,
                              void* d_out, int out_size, void* d_ws, size_t ws_size,
                              hipStream_t stream) {
  const float* x      = (const float*)d_in[0];  // [2,2048,1024]
  const float* w_qkv  = (const float*)d_in[1];  // [1024,3072]
  const float* b_qkv  = (const float*)d_in[2];  // [3072]
  const float* w_proj = (const float*)d_in[3];  // [1024,1024]
  const float* b_proj = (const float*)d_in[4];  // [1024]
  float* out = (float*)d_out;                   // [2,2048,1024]

  char* ws = (char*)d_ws;
  size_t off = 0;
  auto salloc = [&](size_t bytes) { void* p = ws + off; off += (bytes + 255) & ~(size_t)255; return p; };

  u16* x_bf    = (u16*)salloc((size_t)4096 * 1024 * 2);
  u16* wqkv_t  = (u16*)salloc((size_t)3072 * 1024 * 2);
  u16* wproj_t = (u16*)salloc((size_t)1024 * 1024 * 2);
  u16* q_bf    = (u16*)salloc((size_t)32 * 2048 * 64 * 2);
  u16* k_bf    = (u16*)salloc((size_t)32 * 2048 * 64 * 2);
  u16* vt_bf   = (u16*)salloc((size_t)32 * 64 * 2048 * 2);
  u16* ctx_bf  = (u16*)salloc((size_t)4096 * 1024 * 2);

  cvt_bf16_kernel<<<4096, 256, 0, stream>>>(x, x_bf, 4096 * 1024);
  transp_bf16_kernel<<<4096, 256, 0, stream>>>(w_qkv, wqkv_t, 1024, 3072);
  transp_bf16_kernel<<<2048, 256, 0, stream>>>(w_proj, wproj_t, 1024, 1024);

  // QKV projection: M=4096, K=1024, Nout=3072 -> 64*24 tiles
  gemm_bf16_kernel<0><<<dim3(64 * 24), dim3(256), 0, stream>>>(
      x_bf, wqkv_t, b_qkv, nullptr, q_bf, k_bf, vt_bf, 4096, 1024, 3072);

  // attention: 32 heads * 16 query blocks
  flash_attn_kernel<<<dim3(32 * 16), dim3(256), 0, stream>>>(q_bf, k_bf, vt_bf, ctx_bf);

  // output projection: M=4096, K=1024, Nout=1024 -> 64*8 tiles
  gemm_bf16_kernel<1><<<dim3(64 * 8), dim3(256), 0, stream>>>(
      ctx_bf, wproj_t, b_proj, out, nullptr, nullptr, nullptr, 4096, 1024, 1024);
}
